// CNNGenerator_49813030699168
// MI455X (gfx1250) — matrix operand order repaired and checked by reading, not compiled, not run on hardware
//
#include <hip/hip_runtime.h>
#include <hip/hip_bf16.h>
#include <math.h>

// MI455X / gfx1250, wave32. All matmul work mapped to v_wmma_f32_16x16x32_f16.
//
// NOTE on the reference: softmax(a, axis=1) acts on an axis of length 1, so the
// attention weights are identically 1.0 and the aw1..aw5/atw/atb conv stack is
// dead math. final[b,t,c] = sum_{w=0..7} ffc[b, t+w-3, c] (width-8 sliding sum).
//
// Pipeline:
//   K0 pack_weights : conv/FC weights -> per-lane WMMA B-fragment order (f16)
//   K1 cnn_wmma     : per wave: 16 consecutive t samples as the M=16 tile;
//                     conv1..conv4 + FC1..FC3 as chained WMMA GEMMs.
//                     FC-chain activations are stored in A-fragment-swizzled
//                     LDS order so each A fragment loads as one 32B ds access.
//   K2 attn_final   : sliding sum of 8 over T + [32->16] matmul + clip

typedef _Float16 h16;
typedef __attribute__((ext_vector_type(16))) _Float16 v16h;
typedef __attribute__((ext_vector_type(8)))  float    v8f;

#define T_LEN 8192
#define B_SZ  16

// ---- WMMA wave32 fragment element maps (ISA 7.12.2) ----
// A (16x32 f16): element e of lane L: m = L&15,
//   k = ((e&8)<<1) | ((L&16)>>1) | (e&7)
// Inverse (element (m,k) -> lane/slot): lane = m + ((k&8)<<1),
//   e = (k&7) | ((k&16)>>1)
// B (32x16 f16): element e of lane L: n = L&15, k = same map as A
__device__ __forceinline__ int a_klocal(int lane, int e) {
  return ((e & 8) << 1) | ((lane & 16) >> 1) | (e & 7);
}

// offset (in halves) of element (m, ch) in an A-fragment-swizzled activation
// buffer: fragment kt = ch>>5 occupies 512 halves, lane slice = 16 halves.
__device__ __forceinline__ int a_swz_off(int m, int ch) {
  return ((ch >> 5) << 9) + ((m + ((ch & 8) << 1)) << 4) +
         ((ch & 7) | ((ch & 16) >> 1));
}

__device__ __forceinline__ v8f wmma_step(v16h a, v16h b, v8f c) {
  return __builtin_amdgcn_wmma_f32_16x16x32_f16(false, a, false, b, (short)0, c,
                                                false, false);
}

__device__ __forceinline__ v16h load_b(const h16* __restrict__ packB, int f, int lane) {
  // one coalesced 32B/lane load; fragment = 512 halves, lane slice = 16 halves
  return *(const v16h*)(packB + f * 512 + lane * 16);
}

// ---------------- Kernel 0: pack weights into B-fragment order ----------------
// Fragment ids: L1:[0,6) kt3 nt2 | L2:[6,12) | L3:[12,24) kt3 nt4 |
//               L4:[24,48) kt6 nt4 | FC1:[48,64) kt2 nt8 | FC2:[64,80) kt4 nt4 |
//               FC3:[80,84) kt2 nt2.  84 frags * 512 halves = 84KB.
__global__ void pack_weights(const float* __restrict__ w1, const float* __restrict__ w2,
                             const float* __restrict__ w3, const float* __restrict__ w4,
                             const float* __restrict__ fw1, const float* __restrict__ fw2,
                             const float* __restrict__ fw3, h16* __restrict__ packB) {
  for (int idx = threadIdx.x; idx < 84 * 512; idx += blockDim.x) {
    int f = idx >> 9;
    int r = idx & 511;
    int lane = r >> 4;
    int e = r & 15;
    int klocal = ((e & 8) << 1) | ((lane & 16) >> 1) | (e & 7);   // B-fragment k map
    int n = lane & 15;
    const float* w = nullptr;
    int inC = 0, inK = 0, kt = 0, nt = 0, kreal = 0, conv = 0;
    if (f < 6)       { int l = f;      kt = l % 3; nt = l / 3; w = w1;  inC = 29; kreal = 87;  conv = 1; }
    else if (f < 12) { int l = f - 6;  kt = l % 3; nt = l / 3; w = w2;  inC = 32; kreal = 96;  conv = 1; }
    else if (f < 24) { int l = f - 12; kt = l % 3; nt = l / 3; w = w3;  inC = 32; kreal = 96;  conv = 1; }
    else if (f < 48) { int l = f - 24; kt = l % 6; nt = l / 6; w = w4;  inC = 64; kreal = 192; conv = 1; }
    else if (f < 64) { int l = f - 48; kt = l % 2; nt = l / 2; w = fw1; inK = 64;  conv = 0; }
    else if (f < 80) { int l = f - 64; kt = l % 4; nt = l / 4; w = fw2; inK = 128; conv = 0; }
    else             { int l = f - 80; kt = l % 2; nt = l / 2; w = fw3; inK = 64;  conv = 0; }
    int k  = kt * 32 + klocal;
    int ng = nt * 16 + n;
    float v = 0.f;
    if (conv) { if (k < kreal) v = w[(ng * inC + k / 3) * 3 + (k % 3)]; }
    else      { v = w[ng * inK + k]; }
    packB[idx] = (h16)v;
  }
}

// ---------------- A-fragment gathers from wave-private LDS --------------------
__device__ __forceinline__ v16h gather_l1(const h16* __restrict__ win, int lane,
                                          int kt, int p) {
  v16h a; int m = lane & 15;
#pragma unroll
  for (int e = 0; e < 16; ++e) {
    int k = kt * 32 + a_klocal(lane, e);
    int c = k / 3, tau = k - c * 3;
    int j = 2 * p - 1 + tau;                 // conv pad=1, stride=2; j in [-1,15]
    a[e] = (c < 29 && j >= 0) ? win[c * 32 + (m + j)] : (h16)0;
  }
  return a;
}

__device__ __forceinline__ v16h gather_conv(const h16* __restrict__ act, int lane,
                                            int kt, int p, int inC, int P) {
  v16h a; int m = lane & 15;
#pragma unroll
  for (int e = 0; e < 16; ++e) {
    int k = kt * 32 + a_klocal(lane, e);
    int c = k / 3, tau = k - c * 3;
    int j = 2 * p - 1 + tau;
    a[e] = (c < inC && j >= 0 && j < P) ? act[(m * inC + c) * P + j] : (h16)0;
  }
  return a;
}

// FC-chain activations are stored pre-swizzled: one contiguous 32B LDS load.
__device__ __forceinline__ v16h gather_fc_swz(const h16* __restrict__ act, int lane,
                                              int kt) {
  return *(const v16h*)(act + (kt << 9) + (lane << 4));
}

// leaky + bias on f32 accumulator, store as f16 activation [m][stride]
__device__ __forceinline__ void store_act(h16* __restrict__ dst, int stride, int chOff,
                                          v8f acc, float bias, float slope, int mb) {
#pragma unroll
  for (int r = 0; r < 8; ++r) {
    float v = acc[r] + bias;
    v = (v >= 0.f) ? v : slope * v;
    dst[(r + mb) * stride + chOff] = (h16)v;
  }
}

// leaky + bias, store into A-fragment-swizzled order (ch uniform per call, so
// the 8 stores are base + m*16 -- constant-stride LDS stores)
__device__ __forceinline__ void store_act_swz(h16* __restrict__ dst, int ch,
                                              v8f acc, float bias, float slope,
                                              int mb) {
#pragma unroll
  for (int r = 0; r < 8; ++r) {
    float v = acc[r] + bias;
    v = (v >= 0.f) ? v : slope * v;
    dst[a_swz_off(r + mb, ch)] = (h16)v;
  }
}

// ---------------- Kernel 1: conv+MLP, all GEMMs via WMMA ----------------------
__global__ __launch_bounds__(128) void cnn_wmma(
    const float* __restrict__ speech,
    const float* __restrict__ b1, const float* __restrict__ b2,
    const float* __restrict__ b3, const float* __restrict__ b4,
    const float* __restrict__ fb1, const float* __restrict__ fb2,
    const float* __restrict__ fb3,
    const h16* __restrict__ packB, float* __restrict__ ffc) {
  __shared__ h16 s_win[4][29 * 32];                 // per-wave 29 x 31 window
  __shared__ __align__(32) h16 s_act[4][6144];      // ping-pong (A:4096 B:2048)

  const int lane = threadIdx.x & 31;
  const int wv   = threadIdx.x >> 5;
  const int g    = blockIdx.x * 4 + wv;   // 8192 groups of 16 samples
  const int bIdx = g >> 9;                // 512 groups per batch row
  const int t0   = (g & 511) << 4;

  h16* win  = s_win[wv];
  h16* bufA = s_act[wv];
  h16* bufB = s_act[wv] + 4096;
  const int n  = lane & 15;
  const int mb = (lane & 16) >> 1;        // D-tile row base: lanes16-31 hold M=8..15

  // speculative prefetch of this wave's speech stripe (global_prefetch_b8)
  __builtin_prefetch(speech + (size_t)(bIdx * 29) * T_LEN + t0, 0, 1);

  // load window: padded time cols t0-7 .. t0+23 (31 cols), f32 -> f16
  for (int i = lane; i < 29 * 31; i += 32) {
    int c = i / 31, u = i % 31;
    int t = t0 + u - 7;
    float v = (t >= 0 && t < T_LEN) ? speech[(size_t)(bIdx * 29 + c) * T_LEN + t] : 0.f;
    win[c * 32 + u] = (h16)v;
  }
  // LDS regions are wave-private; same-wave DS ops are in-order (no barrier).

  // ---- conv1: 29ch x16 -> 32ch x8  (K=87->96, N=32), leaky 0.02 ----
  for (int p = 0; p < 8; ++p) {
    v8f a0 = {}, a1 = {};
#pragma unroll
    for (int kt = 0; kt < 3; ++kt) {
      v16h av = gather_l1(win, lane, kt, p);
      a0 = wmma_step(av, load_b(packB, 0 + kt, lane), a0);
      a1 = wmma_step(av, load_b(packB, 3 + kt, lane), a1);
    }
    store_act(bufA, 256, n * 8 + p,        a0, b1[n],      0.02f, mb);
    store_act(bufA, 256, (16 + n) * 8 + p, a1, b1[16 + n], 0.02f, mb);
  }

  // ---- conv2: 32x8 -> 32x4 (K=96, N=32), leaky 0.02 ----
  for (int p = 0; p < 4; ++p) {
    v8f a0 = {}, a1 = {};
#pragma unroll
    for (int kt = 0; kt < 3; ++kt) {
      v16h av = gather_conv(bufA, lane, kt, p, 32, 8);
      a0 = wmma_step(av, load_b(packB, 6 + kt, lane), a0);
      a1 = wmma_step(av, load_b(packB, 9 + kt, lane), a1);
    }
    store_act(bufB, 128, n * 4 + p,        a0, b2[n],      0.02f, mb);
    store_act(bufB, 128, (16 + n) * 4 + p, a1, b2[16 + n], 0.02f, mb);
  }

  // ---- conv3: 32x4 -> 64x2 (K=96, N=64), leaky 0.2 ----
  for (int p = 0; p < 2; ++p) {
    v8f acc[4] = {};
#pragma unroll
    for (int kt = 0; kt < 3; ++kt) {
      v16h av = gather_conv(bufB, lane, kt, p, 32, 4);
#pragma unroll
      for (int nt = 0; nt < 4; ++nt)
        acc[nt] = wmma_step(av, load_b(packB, 12 + nt * 3 + kt, lane), acc[nt]);
    }
#pragma unroll
    for (int nt = 0; nt < 4; ++nt)
      store_act(bufA, 128, (nt * 16 + n) * 2 + p, acc[nt], b3[nt * 16 + n], 0.2f, mb);
  }

  // ---- conv4: 64x2 -> 64x1 (K=192, N=64), leaky 0.2; out pre-swizzled ----
  {
    v8f acc[4] = {};
#pragma unroll
    for (int kt = 0; kt < 6; ++kt) {
      v16h av = gather_conv(bufA, lane, kt, 0, 64, 2);
#pragma unroll
      for (int nt = 0; nt < 4; ++nt)
        acc[nt] = wmma_step(av, load_b(packB, 24 + nt * 6 + kt, lane), acc[nt]);
    }
#pragma unroll
    for (int nt = 0; nt < 4; ++nt)
      store_act_swz(bufB, nt * 16 + n, acc[nt], b4[nt * 16 + n], 0.2f, mb);
  }

  // ---- FC1: 64 -> 128, leaky 0.02; in/out pre-swizzled ----
  {
    v8f acc[8] = {};
#pragma unroll
    for (int kt = 0; kt < 2; ++kt) {
      v16h av = gather_fc_swz(bufB, lane, kt);
#pragma unroll
      for (int nt = 0; nt < 8; ++nt)
        acc[nt] = wmma_step(av, load_b(packB, 48 + nt * 2 + kt, lane), acc[nt]);
    }
#pragma unroll
    for (int nt = 0; nt < 8; ++nt)
      store_act_swz(bufA, nt * 16 + n, acc[nt], fb1[nt * 16 + n], 0.02f, mb);
  }

  // ---- FC2: 128 -> 64, leaky 0.02; in/out pre-swizzled ----
  {
    v8f acc[4] = {};
#pragma unroll
    for (int kt = 0; kt < 4; ++kt) {
      v16h av = gather_fc_swz(bufA, lane, kt);
#pragma unroll
      for (int nt = 0; nt < 4; ++nt)
        acc[nt] = wmma_step(av, load_b(packB, 64 + nt * 4 + kt, lane), acc[nt]);
    }
#pragma unroll
    for (int nt = 0; nt < 4; ++nt)
      store_act_swz(bufB, nt * 16 + n, acc[nt], fb2[nt * 16 + n], 0.02f, mb);
  }

  // ---- FC3: 64 -> 32, tanh, write ffc (f32) to workspace ----
  {
    v8f acc[2] = {};
#pragma unroll
    for (int kt = 0; kt < 2; ++kt) {
      v16h av = gather_fc_swz(bufB, lane, kt);
#pragma unroll
      for (int nt = 0; nt < 2; ++nt)
        acc[nt] = wmma_step(av, load_b(packB, 80 + nt * 2 + kt, lane), acc[nt]);
    }
#pragma unroll
    for (int nt = 0; nt < 2; ++nt) {
      int ch = nt * 16 + n;
      float bias = fb3[ch];
#pragma unroll
      for (int r = 0; r < 8; ++r) {
        int m = r + mb;
        ffc[(size_t)(bIdx * T_LEN + t0 + m) * 32 + ch] = tanhf(acc[nt][r] + bias);
      }
    }
  }
}

// ---------------- Kernel 2: sliding sum(8) + [32->16] matmul + clip ----------
__global__ __launch_bounds__(128) void attn_final(
    const float* __restrict__ ffc, const float* __restrict__ fcw,
    float* __restrict__ out) {
  __shared__ float s_f[71 * 32];
  __shared__ float s_sum[64 * 32];
  const int chunk  = blockIdx.x;        // 2048 chunks of 64 t
  const int bIdx   = chunk >> 7;        // 128 chunks per batch row
  const int tstart = (chunk & 127) << 6;

  for (int i = threadIdx.x; i < 71 * 32; i += 128) {
    int row = i >> 5, c = i & 31;
    int t = tstart + row - 3;           // window offsets -3 .. +4
    s_f[i] = (t >= 0 && t < T_LEN) ? ffc[(size_t)(bIdx * T_LEN + t) * 32 + c] : 0.f;
  }
  __syncthreads();
  for (int i = threadIdx.x; i < 64 * 32; i += 128) {
    int tt = i >> 5, c = i & 31;
    float s = 0.f;
#pragma unroll
    for (int w = 0; w < 8; ++w) s += s_f[(tt + w) * 32 + c];
    s_sum[i] = s;
  }
  __syncthreads();
  for (int i = threadIdx.x; i < 64 * 16; i += 128) {
    int tt = i >> 4, k = i & 15;
    float s = 0.f;
#pragma unroll
    for (int c = 0; c < 32; ++c) s += s_sum[tt * 32 + c] * fcw[k * 32 + c];
    s = fminf(fmaxf(s, 0.f), 1.f);
    out[(size_t)(bIdx * T_LEN + tstart + tt) * 16 + k] = s;
  }
}

extern "C" void kernel_launch(void* const* d_in, const int* in_sizes, int n_in,
                              void* d_out, int out_size, void* d_ws, size_t ws_size,
                              hipStream_t stream) {
  const float* speech = (const float*)d_in[0];
  const float* w1  = (const float*)d_in[1];
  const float* b1  = (const float*)d_in[2];
  const float* w2  = (const float*)d_in[3];
  const float* b2  = (const float*)d_in[4];
  const float* w3  = (const float*)d_in[5];
  const float* b3  = (const float*)d_in[6];
  const float* w4  = (const float*)d_in[7];
  const float* b4  = (const float*)d_in[8];
  const float* fw1 = (const float*)d_in[9];
  const float* fb1 = (const float*)d_in[10];
  const float* fw2 = (const float*)d_in[11];
  const float* fb2 = (const float*)d_in[12];
  const float* fw3 = (const float*)d_in[13];
  const float* fb3 = (const float*)d_in[14];
  // d_in[15..26]: aw1..ab5, atw, atb -- dead math (softmax over singleton axis)
  const float* fcw = (const float*)d_in[27];
  float* out = (float*)d_out;

  // workspace: [0,128KB) packed B-fragments (f16), [128KB, +16MB) ffc (f32)
  h16*   packB = (h16*)d_ws;
  float* ffc   = (float*)((char*)d_ws + 131072);

  pack_weights<<<1, 256, 0, stream>>>(w1, w2, w3, w4, fw1, fw2, fw3, packB);
  cnn_wmma<<<2048, 128, 0, stream>>>(speech, b1, b2, b3, b4, fb1, fb2, fb3,
                                     packB, ffc);
  attn_final<<<2048, 128, 0, stream>>>(ffc, fcw, out);
}
